// BrainNetROIEncoder_34806414966789
// MI455X (gfx1250) — compile-verified, matrix operand order checked by reading, not run on hardware
//
#include <hip/hip_runtime.h>

// ---------------------------------------------------------------------------
// BrainNet ROI encoder, CDNA5 (gfx1250) implementation.
//
// Pipeline:
//   1) sel_kernel          : per-batch 4-pass radix select -> threshold key T_b
//                            of the K=3990-th largest strict-upper correlation.
//   2) gcn_kernel          : per-batch (one workgroup):
//                              dis[i] = rsqrt(1 + #{j!=i : key(C[i,j])>=T})
//                              h  = C @ W1                (WMMA f16, f32 acc)
//                              x1 = tanh(M @ h + b1)      (M built on the fly)
//                              h2 = x1 @ W2
//                              x2 = tanh(M @ h2 + b2)
//                              pooled = [mean(x1), mean(x2)]
//                            where M = D^-1/2 (Adj + I) D^-1/2 == the exact
//                            GCNConv propagation of the reference.
//   3) apack/pack kernels  : pack triu(C) per batch to f16 (K padded to 80224),
//                            fold tf-BatchNorm scale into W0 -> f16 W0s in
//                            WMMA B-fragment order; pack W1/W2 likewise.
//   4) mlp_gemm_kernel     : [256 x 80224]f16 @ [80224 x 64]f16 -> f32.
//   5) constvec kernels    : batch-independent BN-fold constant (deterministic
//                            two-stage reduction, no float atomics).
//   6) finalize_kernel     : + pooled-branch contribution, BN0..BN2, tiny MLPs,
//                            projection + ReLU -> d_out [256,256] f32.
//
// WMMA fragment layout assumptions (CDNA5 ISA 7.12.2, wave32):
//   A (16x32 f16): lane L<16 -> row L, element e -> K = (e<8 ? e : e+8);
//                  lane L>=16 -> row L-16, K = 8 + (e<8 ? e : e+8).
//   B (32x16 f16): same indexing with N in place of M (lane = column).
//   C/D (16x16 f32, 8 VGPRs): element v -> M = v + (lane>=16 ? 8 : 0),
//                  N = lane & 15.
// ---------------------------------------------------------------------------

#define N_ROIS   400
#define NBATCH   256
#define KEDGES   3990u
#define TRIU0    80200
#define KPAD     80224          // 2507 * 32
#define KC_MLP   2507
#define CV_BLKS  314            // ceil(80200/256)
#define EPSF     1e-5f

typedef __attribute__((ext_vector_type(16))) _Float16 v16h;
typedef __attribute__((ext_vector_type(8)))  _Float16 v8h;
typedef __attribute__((ext_vector_type(8)))  float    v8f;

__device__ __forceinline__ unsigned fkey(float f) {
  unsigned u = __float_as_uint(f);
  return (u & 0x80000000u) ? ~u : (u | 0x80000000u);   // monotone float->uint
}

__device__ __forceinline__ v8f wmma_f16(v16h a, v16h b, v8f c) {
  return __builtin_amdgcn_wmma_f32_16x16x32_f16(false, a, false, b,
                                                (short)0, c, false, false);
}

// ---------------------------------------------------------------------------
// 1) per-batch radix select: threshold key of K-th largest strict-upper value
// ---------------------------------------------------------------------------
__global__ void __launch_bounds__(256) sel_kernel(const float* __restrict__ x,
                                                  unsigned* __restrict__ Tkey) {
  const int tid = threadIdx.x;
  const float* C = x + (size_t)blockIdx.x * (N_ROIS * N_ROIS);
  __shared__ unsigned hist[256];
  __shared__ unsigned sh_pref, sh_k;

  unsigned pref = 0u, mask = 0u, kneed = KEDGES;
  for (int shift = 24; shift >= 0; shift -= 8) {
    if (tid < 256) hist[tid] = 0u;
    __syncthreads();
    for (int idx = tid; idx < N_ROIS * N_ROIS; idx += 256) {
      int r = idx / N_ROIS;
      int c = idx - r * N_ROIS;
      if (r < c) {
        unsigned u = fkey(C[idx]);
        if ((u & mask) == pref) atomicAdd(&hist[(u >> shift) & 255u], 1u);
      }
    }
    __syncthreads();
    if (tid == 0) {
      unsigned above = 0u; int v = 255;
      for (; v > 0; --v) {
        unsigned h = hist[v];
        if (above + h >= kneed) break;
        above += h;
      }
      sh_pref = pref | ((unsigned)v << shift);
      sh_k    = kneed - above;
    }
    __syncthreads();
    pref  = sh_pref;
    kneed = sh_k;
    mask |= (0xFFu << shift);
    __syncthreads();
  }
  if (tid == 0) Tkey[blockIdx.x] = pref;   // select := key >= pref
}

// ---------------------------------------------------------------------------
// weight packing into WMMA B-fragment order
// ---------------------------------------------------------------------------
__global__ void pack_w1w2_kernel(const float* __restrict__ W1,
                                 const float* __restrict__ W2,
                                 _Float16* __restrict__ W1h,
                                 _Float16* __restrict__ W2h) {
  int idx = blockIdx.x * blockDim.x + threadIdx.x;
  if (idx < 13 * 2 * 32 * 16) {                      // W1: 13 kc x 2 nt
    int e = idx & 15, lane = (idx >> 4) & 31, nt = (idx >> 9) & 1, kc = idx >> 10;
    int k = kc * 32 + ((lane & 16) ? 8 : 0) + (e < 8 ? e : e + 8);
    int n = nt * 16 + (lane & 15);
    W1h[idx] = (_Float16)((k < N_ROIS) ? W1[k * 32 + n] : 0.f);
  }
  int j = idx - 13 * 2 * 32 * 16;
  if (j >= 0 && j < 2 * 32 * 16) {                   // W2: 1 kc x 2 nt
    int e = j & 15, lane = (j >> 4) & 31, nt = (j >> 9) & 1;
    int k = ((lane & 16) ? 8 : 0) + (e < 8 ? e : e + 8);
    int n = nt * 16 + (lane & 15);
    W2h[j] = (_Float16)W2[k * 32 + n];
  }
}

__global__ void pack_w0s_kernel(const float* __restrict__ W0,
                                const float* __restrict__ bn_g,
                                const float* __restrict__ bn_v,
                                _Float16* __restrict__ W0s) {
  int idx = blockIdx.x * blockDim.x + threadIdx.x;
  if (idx >= KC_MLP * 4 * 32 * 16) return;
  int e = idx & 15, lane = (idx >> 4) & 31, nt = (idx >> 9) & 3, kc = idx >> 11;
  int k = kc * 32 + ((lane & 16) ? 8 : 0) + (e < 8 ? e : e + 8);
  int n = nt * 16 + (lane & 15);
  float val = 0.f;
  if (k < TRIU0) {
    float s = bn_g[k] * rsqrtf(bn_v[k] + EPSF);      // fold BN scale into W0
    val = s * W0[(size_t)k * 64 + n];
  }
  W0s[idx] = (_Float16)val;
}

// pack inclusive upper triangle of C (row-major, per batch) to f16, K padded
__global__ void apack_kernel(const float* __restrict__ x,
                             _Float16* __restrict__ Apack) {
  int bid = blockIdx.x;
  int b = bid / N_ROIS, r = bid - b * N_ROIS;
  size_t off0 = (size_t)r * N_ROIS - (size_t)r * (r - 1) / 2; // triu0 row start
  const float* src = x + (size_t)b * (N_ROIS * N_ROIS) + (size_t)r * N_ROIS + r;
  _Float16* dst = Apack + (size_t)b * KPAD + off0;
  int len = N_ROIS - r;
  for (int t = threadIdx.x; t < len; t += blockDim.x) dst[t] = (_Float16)src[t];
  if (r == 0) {   // zero K padding once per batch
    _Float16* pad = Apack + (size_t)b * KPAD + TRIU0;
    for (int t = threadIdx.x; t < KPAD - TRIU0; t += blockDim.x)
      pad[t] = (_Float16)0.f;
  }
}

// ---------------------------------------------------------------------------
// 2) per-batch GCN (one workgroup, 8 waves): dense WMMA with on-the-fly M
// ---------------------------------------------------------------------------
__device__ __forceinline__ void gcn_prop_layer(
    const float* __restrict__ C, unsigned T, const float* dis_s,
    const _Float16* srcbuf,                    // LDS [416][32] f16 (rows>=400 zero)
    const float* __restrict__ bias,            // [32]
    _Float16* storebuf,                        // LDS [416][32] or nullptr
    float* wsum0, float* wsum1,                // LDS [8*32] each
    int wave, int lane) {
  const int c0 = lane & 15;
  const int off = (lane & 16) ? 8 : 0;
  float msum0 = 0.f, msum1 = 0.f;
  for (int rt = wave; rt < 25; rt += 8) {
    v8f acc0 = {}; v8f acc1 = {};
    const int i = rt * 16 + c0;
    const float* Crow = C + (size_t)i * N_ROIS;
    const float disi = dis_s[i];
    for (int kc = 0; kc < 13; ++kc) {
      const int kbase = kc * 32 + off;
      v16h a, b0, b1;
#pragma unroll
      for (int e = 0; e < 16; ++e) {
        int k = kbase + (e < 8 ? e : e + 8);
        float mv = 0.f;
        if (k < N_ROIS) {
          if (k == i) mv = disi * disi;                  // self-loop term
          else if (fkey(Crow[k]) >= T) mv = disi * dis_s[k];
        }
        a[e]  = (_Float16)mv;
        b0[e] = srcbuf[k * 32 + c0];
        b1[e] = srcbuf[k * 32 + 16 + c0];
      }
      acc0 = wmma_f16(a, b0, acc0);
      acc1 = wmma_f16(a, b1, acc1);
    }
#pragma unroll
    for (int v = 0; v < 8; ++v) {
      int m = rt * 16 + v + off;
      float v0 = tanhf(acc0[v] + bias[c0]);
      float v1 = tanhf(acc1[v] + bias[16 + c0]);
      if (storebuf) {
        storebuf[m * 32 + c0]      = (_Float16)v0;
        storebuf[m * 32 + 16 + c0] = (_Float16)v1;
      }
      msum0 += v0; msum1 += v1;
    }
  }
  wsum0[wave * 32 + lane] = msum0;
  wsum1[wave * 32 + lane] = msum1;
}

__global__ void __launch_bounds__(256) gcn_kernel(
    const float* __restrict__ x, const unsigned* __restrict__ Tkey,
    const _Float16* __restrict__ W1h, const _Float16* __restrict__ W2h,
    const float* __restrict__ b1, const float* __restrict__ b2,
    float* __restrict__ pooled) {
  __shared__ float    dis_s[N_ROIS];
  __shared__ _Float16 hbuf[416 * 32];
  __shared__ _Float16 xbuf[416 * 32];
  __shared__ float    wsumA[8 * 32];
  __shared__ float    wsumB[8 * 32];

  const int tid = threadIdx.x, lane = tid & 31, wave = tid >> 5;
  const int b = blockIdx.x;
  const float* C = x + (size_t)b * (N_ROIS * N_ROIS);
  const unsigned T = Tkey[b];
  const int c0 = lane & 15;
  const int off = (lane & 16) ? 8 : 0;

  // ---- phase 0: zero pad rows, degrees -> dis = rsqrt(1 + deg) ----
  for (int i = tid; i < 16 * 32; i += 256) {
    hbuf[N_ROIS * 32 + i] = (_Float16)0.f;
    xbuf[N_ROIS * 32 + i] = (_Float16)0.f;
  }
  for (int r = wave; r < N_ROIS; r += 8) {
    const float* row = C + (size_t)r * N_ROIS;
    int cnt = 0;
    for (int c = lane; c < N_ROIS; c += 32)
      if (c != r && fkey(row[c]) >= T) cnt++;
    for (int o = 16; o; o >>= 1) cnt += __shfl_xor(cnt, o);
    if (lane == 0) dis_s[r] = rsqrtf((float)(1 + cnt));
  }
  __syncthreads();

  // ---- phase A: h = C @ W1  (f16 WMMA, f32 acc), raw (bias/tanh later) ----
  for (int rt = wave; rt < 25; rt += 8) {
    v8f acc0 = {}; v8f acc1 = {};
    const int i = rt * 16 + c0;
    const float* Crow = C + (size_t)i * N_ROIS;
    for (int kc = 0; kc < 13; ++kc) {
      const int kbase = kc * 32 + off;
      v16h a;
#pragma unroll
      for (int e = 0; e < 8; ++e) a[e] = (_Float16)Crow[kbase + e];
      if (kc < 12) {
#pragma unroll
        for (int e = 0; e < 8; ++e) a[8 + e] = (_Float16)Crow[kbase + 16 + e];
      } else {
#pragma unroll
        for (int e = 0; e < 8; ++e) a[8 + e] = (_Float16)0.f;
      }
      v16h w0 = *(const v16h*)(W1h + (size_t)((kc * 2 + 0) * 32 + lane) * 16);
      v16h w1 = *(const v16h*)(W1h + (size_t)((kc * 2 + 1) * 32 + lane) * 16);
      acc0 = wmma_f16(a, w0, acc0);
      acc1 = wmma_f16(a, w1, acc1);
    }
#pragma unroll
    for (int v = 0; v < 8; ++v) {
      int m = rt * 16 + v + off;
      hbuf[m * 32 + c0]      = (_Float16)acc0[v];
      hbuf[m * 32 + 16 + c0] = (_Float16)acc1[v];
    }
  }
  __syncthreads();

  // ---- phase B: x1 = tanh(M @ h + b1) ----
  gcn_prop_layer(C, T, dis_s, hbuf, b1, xbuf, wsumA, wsumB, wave, lane);
  __syncthreads();

  // deterministic mean reduce, layer 1 -> pooled[b][0:32]
  if (tid < 32) {
    float s = 0.f;
    if (tid < 16)
      for (int w = 0; w < 8; ++w) s += wsumA[w * 32 + tid] + wsumA[w * 32 + tid + 16];
    else
      for (int w = 0; w < 8; ++w) s += wsumB[w * 32 + (tid - 16)] + wsumB[w * 32 + tid];
    pooled[(size_t)b * 64 + tid] = s * (1.f / (float)N_ROIS);
  }

  // ---- phase C: h2 = x1 @ W2 (K=32, single chunk) ----
  {
    v16h w2b0 = *(const v16h*)(W2h + (size_t)(0 * 32 + lane) * 16);
    v16h w2b1 = *(const v16h*)(W2h + (size_t)(1 * 32 + lane) * 16);
    for (int rt = wave; rt < 25; rt += 8) {
      const int i = rt * 16 + c0;
      v16h a;
#pragma unroll
      for (int e = 0; e < 16; ++e) {
        int k = off + (e < 8 ? e : e + 8);
        a[e] = xbuf[i * 32 + k];
      }
      v8f z0 = {}; v8f z1 = {};
      v8f acc0 = wmma_f16(a, w2b0, z0);
      v8f acc1 = wmma_f16(a, w2b1, z1);
#pragma unroll
      for (int v = 0; v < 8; ++v) {
        int m = rt * 16 + v + off;
        hbuf[m * 32 + c0]      = (_Float16)acc0[v];
        hbuf[m * 32 + 16 + c0] = (_Float16)acc1[v];
      }
    }
  }
  __syncthreads();

  // ---- phase D: x2 = tanh(M @ h2 + b2), means only ----
  gcn_prop_layer(C, T, dis_s, hbuf, b2, (_Float16*)nullptr, wsumA, wsumB, wave, lane);
  __syncthreads();

  if (tid < 32) {
    float s = 0.f;
    if (tid < 16)
      for (int w = 0; w < 8; ++w) s += wsumA[w * 32 + tid] + wsumA[w * 32 + tid + 16];
    else
      for (int w = 0; w < 8; ++w) s += wsumB[w * 32 + (tid - 16)] + wsumB[w * 32 + tid];
    pooled[(size_t)b * 64 + 32 + tid] = s * (1.f / (float)N_ROIS);
  }
}

// ---------------------------------------------------------------------------
// 4) big GEMM: gout[256,64] = Apack[256,80224]f16 @ W0s[80224,64]f16
// ---------------------------------------------------------------------------
__global__ void __launch_bounds__(256) mlp_gemm_kernel(
    const _Float16* __restrict__ Apack, const _Float16* __restrict__ W0s,
    float* __restrict__ gout) {
  __shared__ float red[8 * 32 * 8];
  const int tid = threadIdx.x, lane = tid & 31, wave = tid >> 5;
  const int nt = wave & 3, ks = wave >> 2;          // 4 N-tiles x 2 K-splits
  const int m0 = blockIdx.x * 16;
  const int c0 = lane & 15;
  const int off = (lane & 16) ? 8 : 0;
  const _Float16* Arow = Apack + (size_t)(m0 + c0) * KPAD;
  const int kc0 = ks ? (KC_MLP / 2) : 0;
  const int kc1 = ks ? KC_MLP : (KC_MLP / 2);

  v8f acc = {};
  for (int kc = kc0; kc < kc1; ++kc) {
    const int kbase = kc * 32 + off;
    v8h alo = *(const v8h*)(Arow + kbase);
    v8h ahi = *(const v8h*)(Arow + kbase + 16);
    v16h a = __builtin_shufflevector(alo, ahi, 0, 1, 2, 3, 4, 5, 6, 7,
                                     8, 9, 10, 11, 12, 13, 14, 15);
    v16h bf = *(const v16h*)(W0s + ((size_t)(kc * 4 + nt) * 32 + lane) * 16);
    acc = wmma_f16(a, bf, acc);
  }
#pragma unroll
  for (int v = 0; v < 8; ++v) red[(wave * 32 + lane) * 8 + v] = acc[v];
  __syncthreads();
  if (ks == 0) {
#pragma unroll
    for (int v = 0; v < 8; ++v) {
      float s = red[(wave * 32 + lane) * 8 + v] + red[((wave + 4) * 32 + lane) * 8 + v];
      int m = m0 + v + off;
      int n = nt * 16 + c0;
      gout[(size_t)m * 64 + n] = s;
    }
  }
}

// ---------------------------------------------------------------------------
// 5) batch-independent BN-fold constant: cvec[m] = sum_e (beta-mean*s)[e]*W0[e][m]
// ---------------------------------------------------------------------------
__global__ void __launch_bounds__(256) constvec_part_kernel(
    const float* __restrict__ W0, const float* __restrict__ g,
    const float* __restrict__ beta, const float* __restrict__ mean,
    const float* __restrict__ var, float* __restrict__ cpart) {
  __shared__ float coef[256];
  const int e0 = blockIdx.x * 256;
  const int e = e0 + threadIdx.x;
  float cf = 0.f;
  if (e < TRIU0) {
    float s = g[e] * rsqrtf(var[e] + EPSF);
    cf = beta[e] - mean[e] * s;
  }
  coef[threadIdx.x] = cf;
  __syncthreads();
  if (threadIdx.x < 64) {
    int lim = TRIU0 - e0; if (lim > 256) lim = 256;
    float acc = 0.f;
    for (int i = 0; i < lim; ++i)
      acc += coef[i] * W0[(size_t)(e0 + i) * 64 + threadIdx.x];
    cpart[blockIdx.x * 64 + threadIdx.x] = acc;
  }
}

__global__ void __launch_bounds__(64) constvec_reduce_kernel(
    const float* __restrict__ cpart, float* __restrict__ cvec) {
  const int t = threadIdx.x;
  float acc = 0.f;
  for (int bz = 0; bz < CV_BLKS; ++bz) acc += cpart[bz * 64 + t];
  cvec[t] = acc;
}

// ---------------------------------------------------------------------------
// 6) finalize: pooled branch + BN0..BN2 + small MLPs + projection + ReLU
// ---------------------------------------------------------------------------
struct MlpP {
  const float *pooled, *gout, *cvec;
  const float *W0, *b0;
  const float *bhg, *bhb, *bhm, *bhv;
  const float *g0, *be0, *m0, *v0;
  const float *W1m, *b1m;
  const float *g1, *be1, *m1, *v1;
  const float *W2m, *b2m;
  const float *g2, *be2, *m2, *v2;
  const float *Wp, *bp;
  float* out;
};

__global__ void __launch_bounds__(64) finalize_kernel(MlpP P) {
  __shared__ float hbn[64], z0[64], z1[32], z2[32];
  const int b = blockIdx.x, t = threadIdx.x;
  {
    float th = tanhf(P.pooled[(size_t)b * 64 + t]);
    hbn[t] = (th - P.bhm[t]) * (P.bhg[t] * rsqrtf(P.bhv[t] + EPSF)) + P.bhb[t];
  }
  __syncthreads();
  {
    float acc = P.gout[(size_t)b * 64 + t] + P.cvec[t] + P.b0[t];
    for (int j = 0; j < 64; ++j)
      acc += hbn[j] * P.W0[(size_t)(TRIU0 + j) * 64 + t];
    float v = (acc - P.m0[t]) * (P.g0[t] * rsqrtf(P.v0[t] + EPSF)) + P.be0[t];
    z0[t] = fmaxf(v, 0.f);
  }
  __syncthreads();
  if (t < 32) {
    float acc = P.b1m[t];
    for (int m = 0; m < 64; ++m) acc += z0[m] * P.W1m[m * 32 + t];
    float v = (acc - P.m1[t]) * (P.g1[t] * rsqrtf(P.v1[t] + EPSF)) + P.be1[t];
    z1[t] = fmaxf(v, 0.f);
  }
  __syncthreads();
  if (t < 32) {
    float acc = P.b2m[t];
    for (int j = 0; j < 32; ++j) acc += z1[j] * P.W2m[j * 32 + t];
    float v = (acc - P.m2[t]) * (P.g2[t] * rsqrtf(P.v2[t] + EPSF)) + P.be2[t];
    z2[t] = fmaxf(v, 0.f);
  }
  __syncthreads();
  for (int n = t; n < 256; n += 64) {
    float acc = P.bp[n];
    for (int j = 0; j < 32; ++j) acc += z2[j] * P.Wp[j * 256 + n];
    P.out[(size_t)b * 256 + n] = fmaxf(acc, 0.f);
  }
}

// ---------------------------------------------------------------------------
// host
// ---------------------------------------------------------------------------
extern "C" void kernel_launch(void* const* d_in, const int* in_sizes, int n_in,
                              void* d_out, int out_size, void* d_ws, size_t ws_size,
                              hipStream_t stream) {
  (void)in_sizes; (void)n_in; (void)out_size; (void)ws_size;
  // setup_inputs() flattening order (dicts in insertion order, leaves depth-first)
  const float* x     = (const float*)d_in[0];   // [256,400,400]
  const float* W1    = (const float*)d_in[1];   // [400,32]
  const float* b1    = (const float*)d_in[2];   // [32]
  const float* W2    = (const float*)d_in[3];   // [32,32]
  const float* b2    = (const float*)d_in[4];   // [32]
  const float* bng   = (const float*)d_in[5];   // bn.gamma  [80200]
  const float* bnb   = (const float*)d_in[6];   // bn.beta
  const float* bnm   = (const float*)d_in[7];   // bn.mean
  const float* bnv   = (const float*)d_in[8];   // bn.var
  const float* bhg   = (const float*)d_in[9];   // bnh.gamma [64]
  const float* bhb   = (const float*)d_in[10];
  const float* bhm   = (const float*)d_in[11];
  const float* bhv   = (const float*)d_in[12];
  const float* W0    = (const float*)d_in[13];  // mlp0_W [80264,64]
  const float* b0    = (const float*)d_in[14];  // mlp0_b [64]
  const float* g0    = (const float*)d_in[15];  // mlp_bn0.* [64]
  const float* be0   = (const float*)d_in[16];
  const float* m0    = (const float*)d_in[17];
  const float* v0    = (const float*)d_in[18];
  const float* W1m   = (const float*)d_in[19];  // mlp1_W [64,32]
  const float* b1m   = (const float*)d_in[20];
  const float* g1    = (const float*)d_in[21];  // mlp_bn1.* [32]
  const float* be1   = (const float*)d_in[22];
  const float* m1    = (const float*)d_in[23];
  const float* v1    = (const float*)d_in[24];
  const float* W2m   = (const float*)d_in[25];  // mlp2_W [32,32]
  const float* b2m   = (const float*)d_in[26];
  const float* g2    = (const float*)d_in[27];  // mlp_bn2.* [32]
  const float* be2   = (const float*)d_in[28];
  const float* m2    = (const float*)d_in[29];
  const float* v2    = (const float*)d_in[30];
  const float* Wp    = (const float*)d_in[31];  // proj_W [32,256]
  const float* bp    = (const float*)d_in[32];  // proj_b [256]
  float* out = (float*)d_out;                   // [256,256]

  // workspace layout (~52 MB), 256B-aligned slots
  char* ws = (char*)d_ws;
  size_t cur = 0;
  auto take = [&](size_t bytes) -> void* {
    cur = (cur + 255) & ~(size_t)255;
    void* p = ws + cur;
    cur += bytes;
    return p;
  };
  unsigned* Tkey   = (unsigned*)take(sizeof(unsigned) * NBATCH);
  float*    pooled = (float*)   take(sizeof(float) * NBATCH * 64);
  _Float16* W1h    = (_Float16*)take(2ull * 13 * 2 * 32 * 16);
  _Float16* W2h    = (_Float16*)take(2ull * 1 * 2 * 32 * 16);
  _Float16* W0s    = (_Float16*)take(2ull * KC_MLP * 4 * 32 * 16);
  _Float16* Apack  = (_Float16*)take(2ull * NBATCH * KPAD);
  float*    cpart  = (float*)   take(sizeof(float) * CV_BLKS * 64);
  float*    cvec   = (float*)   take(sizeof(float) * 64);
  float*    gout   = (float*)   take(sizeof(float) * NBATCH * 64);

  pack_w1w2_kernel<<<dim3(56), dim3(256), 0, stream>>>(W1, W2, W1h, W2h);
  pack_w0s_kernel<<<dim3((KC_MLP * 4 * 32 * 16 + 255) / 256), dim3(256), 0, stream>>>(
      W0, bng, bnv, W0s);
  apack_kernel<<<dim3(NBATCH * N_ROIS), dim3(128), 0, stream>>>(x, Apack);
  sel_kernel<<<dim3(NBATCH), dim3(256), 0, stream>>>(x, Tkey);
  gcn_kernel<<<dim3(NBATCH), dim3(256), 0, stream>>>(x, Tkey, W1h, W2h, b1, b2, pooled);
  constvec_part_kernel<<<dim3(CV_BLKS), dim3(256), 0, stream>>>(W0, bng, bnb, bnm, bnv, cpart);
  constvec_reduce_kernel<<<dim3(1), dim3(64), 0, stream>>>(cpart, cvec);
  mlp_gemm_kernel<<<dim3(NBATCH / 16), dim3(256), 0, stream>>>(Apack, W0s, gout);

  MlpP P;
  P.pooled = pooled; P.gout = gout; P.cvec = cvec;
  P.W0 = W0; P.b0 = b0;
  P.bhg = bhg; P.bhb = bhb; P.bhm = bhm; P.bhv = bhv;
  P.g0 = g0; P.be0 = be0; P.m0 = m0; P.v0 = v0;
  P.W1m = W1m; P.b1m = b1m;
  P.g1 = g1; P.be1 = be1; P.m1 = m1; P.v1 = v1;
  P.W2m = W2m; P.b2m = b2m;
  P.g2 = g2; P.be2 = be2; P.m2 = m2; P.v2 = v2;
  P.Wp = Wp; P.bp = bp; P.out = out;
  finalize_kernel<<<dim3(NBATCH), dim3(64), 0, stream>>>(P);
}
